// SAGE_16209206575324
// MI455X (gfx1250) — matrix operand order repaired and checked by reading, not compiled, not run on hardware
//
#include <hip/hip_runtime.h>
#include <stdint.h>

// ---------------------------------------------------------------------------
// Types for CDNA5 WMMA (wave32): v_wmma_f32_16x16x32_bf16
// ---------------------------------------------------------------------------
typedef __attribute__((ext_vector_type(16))) __bf16    v16bf;
typedef __attribute__((ext_vector_type(8)))  float     v8f;
typedef __attribute__((ext_vector_type(8)))  unsigned  v8u;

// float -> bf16 (round to nearest even), as ushort
__device__ __forceinline__ unsigned short f2bf(float f) {
    unsigned u = __float_as_uint(f);
    u += 0x7FFFu + ((u >> 16) & 1u);
    return (unsigned short)(u >> 16);
}
__device__ __forceinline__ unsigned f2bf_pk(float lo, float hi) {
    return (unsigned)f2bf(lo) | ((unsigned)f2bf(hi) << 16);
}

// ---------------------------------------------------------------------------
// Degree: deg[dst[e]] += 1.0f   (L2-resident f32 atomics)
// ---------------------------------------------------------------------------
__global__ void degree_kernel(const int* __restrict__ dst, float* __restrict__ deg, int E) {
    int e = blockIdx.x * blockDim.x + threadIdx.x;
    if (e >= E) return;
    atomicAdd(&deg[dst[e]], 1.0f);
}

// ---------------------------------------------------------------------------
// Edge scatter: agg[dst[e], :] += feat[src[e], :]
// One thread per (edge, float4 chunk). C4 is a power of two -> shifts only.
// feat and agg are L2-resident (<=102MB each vs 192MB L2).
// ---------------------------------------------------------------------------
template <int C4>
__global__ void scatter_add_kernel(const float* __restrict__ feat,
                                   const int* __restrict__ src,
                                   const int* __restrict__ dst,
                                   float* __restrict__ agg, int E) {
    long long t = (long long)blockIdx.x * blockDim.x + threadIdx.x;
    if (t >= (long long)E * C4) return;
    int e  = (int)(t / C4);          // constant power-of-two -> shift
    int c4 = (int)(t & (C4 - 1));
    int s = src[e], d = dst[e];
    const float4 v = ((const float4*)(feat))[s * C4 + c4];
    float* o = agg + ((long long)d * C4 + c4) * 4;
    atomicAdd(o + 0, v.x);
    atomicAdd(o + 1, v.y);
    atomicAdd(o + 2, v.z);
    atomicAdd(o + 3, v.w);
}

// ---------------------------------------------------------------------------
// Fragment-layout helpers (CDNA5 ISA 7.12.2, wave32, 16-bit operands)
//  A 16x32 tile: lane = (m%16) + 16*kgrp, kgrp=(kk>>3)&1,
//                vgpr r = ((kk&7)>>1) + 4*(kk>>4), halfword = kk&1
//  B 32x16 tile: lane = (n%16) + 16*((kk>>3)&1), vgpr v = ((kk&7)>>1) + 4*(kk>>4), half = kk&1
// Packed storage: fragment-major, 32 lanes x 8 dwords contiguous per tile.
// ---------------------------------------------------------------------------

// Pack A (bf16, swizzled): k in [0,C) = agg/deg, k in [C,2C) = feat.
// grid.x = N nodes, block = C/2 threads; each thread packs one dword for
// each of the two K-halves.
template <int C>
__global__ void pack_A_kernel(const float* __restrict__ agg,
                              const float* __restrict__ feat,
                              const float* __restrict__ deg,
                              unsigned* __restrict__ Apk /* dwords */) {
    const int K  = 2 * C;
    const int KT = K / 32;
    int n = blockIdx.x;
    int c = threadIdx.x * 2;                 // even channel
    float inv = 1.0f / fmaxf(deg[n], 1.0f);

    // source values
    const float* ar = agg  + (long long)n * C + c;
    const float* fr = feat + (long long)n * C + c;
    unsigned w0 = f2bf_pk(ar[0] * inv, ar[1] * inv);  // k = c, c+1
    unsigned w1 = f2bf_pk(fr[0], fr[1]);              // k = C+c, C+c+1

    int tmBase = (n >> 4) * KT;
    int mrow = n & 15;
#pragma unroll
    for (int half = 0; half < 2; ++half) {
        int k = half ? (C + c) : c;
        unsigned w = half ? w1 : w0;
        int kt = k >> 5, kk = k & 31;
        int kgrp = (kk >> 3) & 1;
        int r = ((kk & 7) >> 1) + ((kk >> 4) << 2);
        int lane = mrow + (kgrp << 4);
        Apk[(((tmBase + kt) << 5) + lane) * 8 + r] = w;
    }
}

// Pack W (bf16, swizzled): Wcat = [Wl ; Wr], shape [2C, NOUT].
template <int C, int NOUT>
__global__ void pack_W_kernel(const float* __restrict__ Wl,
                              const float* __restrict__ Wr,
                              unsigned* __restrict__ Wpk /* dwords */) {
    const int TN = NOUT / 16;
    int t = blockIdx.x * blockDim.x + threadIdx.x;   // over (k/2, j)
    if (t >= C * NOUT) return;                       // 2C*NOUT/2 pairs
    int k = (t / NOUT) * 2;                          // even k
    int j = t & (NOUT - 1);
    const float* Wsrc = (k < C) ? (Wl + k * NOUT + j) : (Wr + (k - C) * NOUT + j);
    unsigned w = f2bf_pk(Wsrc[0], Wsrc[NOUT]);       // k, k+1 same column
    int kt = k >> 5, kk = k & 31;
    int lane = (j & 15) + (((kk >> 3) & 1) << 4);
    int v = ((kk & 7) >> 1) + ((kk >> 4) << 2);
    int tn = j >> 4;
    Wpk[(((kt * TN + tn) << 5) + lane) * 8 + v] = w;
}

// ---------------------------------------------------------------------------
// WMMA GEMM: one wave computes a 16 x NOUT strip.  A fragment loaded once per
// k-step and reused across TN WMMAs (independent accumulator chains -> no
// D->A/B hazards back-to-back).  All fragment loads are contiguous 32B/lane.
// ---------------------------------------------------------------------------
template <int KT, int TN, int RELU, int HAS_PRE>
__global__ void sage_gemm_wmma(const v8u* __restrict__ Apk,
                               const v8u* __restrict__ Wpk,
                               const float* __restrict__ bias,
                               float* __restrict__ out_pre,
                               float* __restrict__ out_post,
                               int Mtiles) {
    const int wave = threadIdx.x >> 5;
    const int lane = threadIdx.x & 31;
    int tm = blockIdx.x * (blockDim.x >> 5) + wave;
    if (tm >= Mtiles) return;                       // per-wave uniform

    const v8u* Ap = Apk + ((tm * KT) << 5) + lane;  // + kt*32 per step
    const v8u* Wp = Wpk + lane;                     // + (kt*TN+tn)*32 per step

    v8f acc[TN];
#pragma unroll
    for (int tn = 0; tn < TN; ++tn) acc[tn] = (v8f)(0.0f);

#pragma unroll
    for (int kt = 0; kt < KT; ++kt) {
        v16bf av = __builtin_bit_cast(v16bf, Ap[kt << 5]);
#pragma unroll
        for (int tn = 0; tn < TN; ++tn) {
            v16bf bv = __builtin_bit_cast(v16bf, Wp[(kt * TN + tn) << 5]);
            acc[tn] = __builtin_amdgcn_wmma_f32_16x16x32_bf16(
                false, av, false, bv, (short)0, acc[tn], false, false);
        }
    }

    const int lane16 = lane & 15;
    const int kgrp = lane >> 4;
    const int Nout = TN * 16;
    // D layout: vgpr r, lane -> m = r + 8*kgrp (in tile), n = lane16
    int base = (tm * 16 + kgrp * 8) * Nout + lane16;   // < 2^31 for all layers
#pragma unroll
    for (int tn = 0; tn < TN; ++tn) {
        float bb = bias[tn * 16 + lane16];
#pragma unroll
        for (int r = 0; r < 8; ++r) {
            float val = acc[tn][r] + bb;
            int idx = base + r * Nout + tn * 16;
            if (HAS_PRE) out_pre[idx] = val;
            out_post[idx] = RELU ? fmaxf(val, 0.0f) : val;
        }
    }
}

// ---------------------------------------------------------------------------
// Host side
// ---------------------------------------------------------------------------
static inline char* alignp(char* p, size_t a) {
    return (char*)(((uintptr_t)p + a - 1) & ~(a - 1));
}

template <int C, int COUT, int RELU, int HAS_PRE>
static void run_layer(const float* feat,
                      const float* Wl, const float* Wr, const float* b,
                      const int* src, const int* dst, int N, int E,
                      float* agg, unsigned* Apk, unsigned* Wpk, const float* deg,
                      float* out_pre, float* out_post, hipStream_t stream) {
    const int KT = (2 * C) / 32;
    const int TN = COUT / 16;

    hipMemsetAsync(agg, 0, (size_t)N * C * sizeof(float), stream);

    long long sc_threads = (long long)E * (C / 4);
    scatter_add_kernel<C / 4><<<(unsigned)((sc_threads + 255) / 256), 256, 0, stream>>>(
        feat, src, dst, agg, E);

    pack_A_kernel<C><<<N, C / 2, 0, stream>>>(agg, feat, deg, Apk);

    int pw_threads = C * COUT;   // one thread per bf16 pair
    pack_W_kernel<C, COUT><<<(pw_threads + 255) / 256, 256, 0, stream>>>(Wl, Wr, Wpk);

    int Mtiles = N / 16;
    sage_gemm_wmma<KT, TN, RELU, HAS_PRE><<<(Mtiles + 7) / 8, 256, 0, stream>>>(
        (const v8u*)Apk, (const v8u*)Wpk, b, out_pre, out_post, Mtiles);
}

extern "C" void kernel_launch(void* const* d_in, const int* in_sizes, int n_in,
                              void* d_out, int out_size, void* d_ws, size_t ws_size,
                              hipStream_t stream) {
    const int IN_C = 256, HID_C = 128, OUT_C = 64;
    const int N = in_sizes[0] / IN_C;       // 100000
    const int E = in_sizes[1] / 2;          // 1600000

    const float* x   = (const float*)d_in[0];
    const int*   src = (const int*)d_in[1];
    const int*   dst = src + E;
    const float* Wl0 = (const float*)d_in[2];
    const float* Wr0 = (const float*)d_in[3];
    const float* b0  = (const float*)d_in[4];
    const float* Wl1 = (const float*)d_in[5];
    const float* Wr1 = (const float*)d_in[6];
    const float* b1  = (const float*)d_in[7];
    const float* Wl2 = (const float*)d_in[8];
    const float* Wr2 = (const float*)d_in[9];
    const float* b2  = (const float*)d_in[10];

    // Output layout: x_final [N,64] | out1 [N,128] | g [N,128]
    float* x_final = (float*)d_out;
    float* out1    = x_final + (size_t)N * OUT_C;
    float* g       = out1 + (size_t)N * HID_C;

    // Workspace carve (~257 MB)
    char* p = (char*)d_ws;
    float* deg = (float*)p;        p = alignp(p + (size_t)N * 4, 256);
    float* agg = (float*)p;        p = alignp(p + (size_t)N * IN_C * 4, 256);
    unsigned* Apk = (unsigned*)p;  p = alignp(p + (size_t)N * 2 * IN_C * 2, 256);
    unsigned* Wpk = (unsigned*)p;  p = alignp(p + (size_t)2 * IN_C * HID_C * 2, 256);
    float* h0 = (float*)p;         p = alignp(p + (size_t)N * HID_C * 4, 256);
    (void)ws_size; (void)n_in; (void)out_size;

    // Degrees (shared by all 3 layers)
    hipMemsetAsync(deg, 0, (size_t)N * sizeof(float), stream);
    degree_kernel<<<(E + 255) / 256, 256, 0, stream>>>(dst, deg, E);

    // Layer 0: x[256] -> h0[128], ReLU
    run_layer<IN_C, HID_C, 1, 0>(x, Wl0, Wr0, b0, src, dst, N, E,
                                 agg, Apk, Wpk, deg, nullptr, h0, stream);
    // Layer 1: h0[128] -> out1 (pre-ReLU) & g (post-ReLU)
    run_layer<HID_C, HID_C, 1, 1>(h0, Wl1, Wr1, b1, src, dst, N, E,
                                  agg, Apk, Wpk, deg, out1, g, stream);
    // Layer 2: g[128] -> x_final[64], no ReLU
    run_layer<HID_C, OUT_C, 0, 0>(g, Wl2, Wr2, b2, src, dst, N, E,
                                  agg, Apk, Wpk, deg, nullptr, x_final, stream);
}
